// ConstrainDotAttention_13709535609369
// MI455X (gfx1250) — matrix operand order repaired and checked by reading, not compiled, not run on hardware
//
#include <hip/hip_runtime.h>

typedef __attribute__((ext_vector_type(16))) _Float16 v16h;
typedef __attribute__((ext_vector_type(8)))  _Float16 v8h;
typedef __attribute__((ext_vector_type(4)))  _Float16 v4h;
typedef __attribute__((ext_vector_type(2)))  _Float16 v2h;
typedef __attribute__((ext_vector_type(8)))  float    v8f;
typedef __attribute__((ext_vector_type(8)))  unsigned v8u;

#define B_   4
#define H_   12
#define S_   2048
#define D_   64
#define QB   128          // queries per workgroup (16 per wave x 8 waves)
#define KT   64           // keys per LDS tile (one softmax update per 64 keys)
#define NKT  (S_ / KT)    // 32 iterations
#define KSTR 72           // padded LDS stride (halves) for K tile rows
#define VSTR 72           // padded LDS stride (halves) for V^T rows (64 keys + pad)

// scale = (1/sqrt(64)) * log2(e); softmax runs in the base-2 domain -> v_exp_f32 directly
#define QSCALE 0.1803368801111204f

// sched_group_barrier masks
#define SG_WMMA   0x008
#define SG_DSREAD 0x100

__global__ __launch_bounds__(256) void fa_fwd_kernel(
    const float* __restrict__ Qg, const float* __restrict__ Kg,
    const float* __restrict__ Vg, const float* __restrict__ Mg,
    float* __restrict__ Og)
{
  __shared__ __align__(16) _Float16 sK[KT * KSTR];   // [key][feat] f16
  __shared__ __align__(16) _Float16 sVt[D_ * VSTR];  // [dim][key]  f16, pre-multiplied by mask[key]

  const int tid  = threadIdx.x;
  const int lane = tid & 31;
  const int wave = tid >> 5;
  const int q    = lane & 15;   // query column owned by this lane
  const int hi   = lane >> 4;   // 0: low half, 1: high half

  const int bh   = blockIdx.y;
  const int qblk = blockIdx.x;

  const float* gK = Kg + (size_t)bh * S_ * D_;
  const float* gV = Vg + (size_t)bh * S_ * D_;
  const float* gM = Mg + (size_t)bh * S_;

  // ---- Q fragment: B-operand of S^T = K x Q^T, pre-scaled (scale * log2e) ----
  v16h qf[2];
  {
    const float* qrow = Qg + (((size_t)bh * S_) + (size_t)qblk * QB + wave * 16 + q) * D_;
    #pragma unroll
    for (int c = 0; c < 2; ++c) {
      const int f0 = c * 32 + hi * 16;
      #pragma unroll
      for (int i = 0; i < 16; ++i)
        qf[c][i] = (_Float16)(qrow[f0 + i] * QSCALE);
    }
  }

  // ---- software-pipelined tile staging (global -> regs -> LDS) ----
  float4 skk[4], svv[4];
  float  smk[4];
  auto stage_loads = [&](int kt) {
    const float4* gk4 = (const float4*)(gK + (size_t)kt * KT * D_);
    const float4* gv4 = (const float4*)(gV + (size_t)kt * KT * D_);
    #pragma unroll
    for (int j = 0; j < 4; ++j) {           // 1024 float4 per tile / 256 threads
      const int idx = tid + j * 256;
      skk[j] = gk4[idx];
      svv[j] = gv4[idx];
      smk[j] = gM[kt * KT + (idx >> 4)];    // L2-resident mask value for this key
    }
  };
  auto store_tile = [&]() {
    #pragma unroll
    for (int j = 0; j < 4; ++j) {
      const int idx = tid + j * 256;
      const int key = idx >> 4;
      const int f4  = (idx & 15) * 4;
      v4h hk = { (_Float16)skk[j].x, (_Float16)skk[j].y,
                 (_Float16)skk[j].z, (_Float16)skk[j].w };
      *(v4h*)&sK[key * KSTR + f4] = hk;
      const float mv = smk[j];              // post-softmax mask folded into V
      sVt[(f4 + 0) * VSTR + key] = (_Float16)(svv[j].x * mv);
      sVt[(f4 + 1) * VSTR + key] = (_Float16)(svv[j].y * mv);
      sVt[(f4 + 2) * VSTR + key] = (_Float16)(svv[j].z * mv);
      sVt[(f4 + 3) * VSTR + key] = (_Float16)(svv[j].w * mv);
    }
  };

  v8f acc[4];
  #pragma unroll
  for (int t = 0; t < 4; ++t)
    #pragma unroll
    for (int e = 0; e < 8; ++e) acc[t][e] = 0.f;
  float m = -1e30f, l = 0.f;

  stage_loads(0);

  for (int kt = 0; kt < NKT; ++kt) {
    __syncthreads();                 // all waves finished reading previous tile
    store_tile();                    // regs -> LDS (f32->f16 convert, mask fold)
    if (kt + 1 < NKT) stage_loads(kt + 1);            // overlap with compute below
    if (kt + 2 < NKT) {                               // warm L1 two tiles ahead
      __builtin_prefetch(gK + (size_t)(kt + 2) * KT * D_ + tid * 16, 0, 3);
      __builtin_prefetch(gV + (size_t)(kt + 2) * KT * D_ + tid * 16, 0, 3);
    }
    __syncthreads();                 // tile visible to all waves

    // Scheduler pipeline: [16 DS reads: ka] -> [8 WMMA: QK] -> [8 DS reads: va0];
    // everything else free so reduction VALU / va loads fill WMMA hazard slots.
    __builtin_amdgcn_sched_group_barrier(SG_DSREAD, 16, 0);
    __builtin_amdgcn_sched_group_barrier(SG_WMMA,    8, 0);
    __builtin_amdgcn_sched_group_barrier(SG_DSREAD,  8, 0);

    // ---- all K fragments (16 b128 DS reads) ----
    v16h ka[8];                      // [keyrow r=0..3][feature chunk ch=0..1]
    #pragma unroll
    for (int r = 0; r < 4; ++r)
      #pragma unroll
      for (int ch = 0; ch < 2; ++ch) {
        const _Float16* kb = &sK[(r * 16 + q) * KSTR + ch * 32 + hi * 16];
        v8h lo  = *(const v8h*)kb;
        v8h h8  = *(const v8h*)(kb + 8);
        ka[r * 2 + ch] = __builtin_shufflevector(lo, h8, 0,1,2,3,4,5,6,7,8,9,10,11,12,13,14,15);
      }

    // ---- S^T tiles: keys [0..15],[16..31],[32..47],[48..63] ----
    v8f c0, c1, c2, c3;
    #pragma unroll
    for (int e = 0; e < 8; ++e) { c0[e] = 0.f; c1[e] = 0.f; c2[e] = 0.f; c3[e] = 0.f; }
    c0 = __builtin_amdgcn_wmma_f32_16x16x32_f16(false, ka[0], false, qf[0], (short)0, c0, false, false);
    c1 = __builtin_amdgcn_wmma_f32_16x16x32_f16(false, ka[2], false, qf[0], (short)0, c1, false, false);
    c2 = __builtin_amdgcn_wmma_f32_16x16x32_f16(false, ka[4], false, qf[0], (short)0, c2, false, false);
    c3 = __builtin_amdgcn_wmma_f32_16x16x32_f16(false, ka[6], false, qf[0], (short)0, c3, false, false);
    c0 = __builtin_amdgcn_wmma_f32_16x16x32_f16(false, ka[1], false, qf[1], (short)0, c0, false, false);
    c1 = __builtin_amdgcn_wmma_f32_16x16x32_f16(false, ka[3], false, qf[1], (short)0, c1, false, false);
    c2 = __builtin_amdgcn_wmma_f32_16x16x32_f16(false, ka[5], false, qf[1], (short)0, c2, false, false);
    c3 = __builtin_amdgcn_wmma_f32_16x16x32_f16(false, ka[7], false, qf[1], (short)0, c3, false, false);

    // V fragments for key-chunk 0 (fill WMMA hazard slots / hide under softmax)
    v16h va0[4];
    #pragma unroll
    for (int t = 0; t < 4; ++t) {
      const _Float16* vb = &sVt[(t * 16 + q) * VSTR + hi * 16];
      v8h lo  = *(const v8h*)vb;
      v8h h8  = *(const v8h*)(vb + 8);
      va0[t] = __builtin_shufflevector(lo, h8, 0,1,2,3,4,5,6,7,8,9,10,11,12,13,14,15);
    }

    // ---- online softmax over 64 keys: balanced-tree max (short dep chain) ----
    float mx[8];
    #pragma unroll
    for (int e = 0; e < 8; ++e)
      mx[e] = fmaxf(fmaxf(c0[e], c1[e]), fmaxf(c2[e], c3[e]));
    mx[0] = fmaxf(mx[0], mx[4]); mx[1] = fmaxf(mx[1], mx[5]);
    mx[2] = fmaxf(mx[2], mx[6]); mx[3] = fmaxf(mx[3], mx[7]);
    mx[0] = fmaxf(mx[0], mx[2]); mx[1] = fmaxf(mx[1], mx[3]);
    float tm = fmaxf(mx[0], mx[1]);
    tm = fmaxf(tm, __shfl_xor(tm, 16, 32));       // partner lane has other 32 keys
    const float mn   = fmaxf(m, tm);
    const float corr = __builtin_amdgcn_exp2f(m - mn);
    m = mn;
    l *= corr;
    #pragma unroll
    for (int t = 0; t < 4; ++t)
      #pragma unroll
      for (int e = 0; e < 8; ++e) acc[t][e] *= corr;

    float p0[8], p1[8], p2[8], p3[8], sp[8];
    #pragma unroll
    for (int e = 0; e < 8; ++e) {
      p0[e] = __builtin_amdgcn_exp2f(c0[e] - m);
      p1[e] = __builtin_amdgcn_exp2f(c1[e] - m);
      p2[e] = __builtin_amdgcn_exp2f(c2[e] - m);
      p3[e] = __builtin_amdgcn_exp2f(c3[e] - m);
      sp[e] = (p0[e] + p1[e]) + (p2[e] + p3[e]);  // parallel across e
    }
    // balanced-tree sum of sp[0..7]
    sp[0] += sp[4]; sp[1] += sp[5]; sp[2] += sp[6]; sp[3] += sp[7];
    sp[0] += sp[2]; sp[1] += sp[3];
    l += sp[0] + sp[1];              // denominator over ALL keys (mask lives in V)

    // V fragments for key-chunk 1 (overlap with pack/exchange below)
    v16h va1[4];
    #pragma unroll
    for (int t = 0; t < 4; ++t) {
      const _Float16* vb = &sVt[(t * 16 + q) * VSTR + 32 + hi * 16];
      v8h lo  = *(const v8h*)vb;
      v8h h8  = *(const v8h*)(vb + 8);
      va1[t] = __builtin_shufflevector(lo, h8, 0,1,2,3,4,5,6,7,8,9,10,11,12,13,14,15);
    }

    // ---- P^T B-fragments: pack f16 pairs, one xor-16 exchange per pair ----
    unsigned u0[4], u1[4], u2[4], u3[4];
    #pragma unroll
    for (int i = 0; i < 4; ++i) {
      v2h t0 = { (_Float16)p0[2 * i], (_Float16)p0[2 * i + 1] };
      v2h t1 = { (_Float16)p1[2 * i], (_Float16)p1[2 * i + 1] };
      v2h t2 = { (_Float16)p2[2 * i], (_Float16)p2[2 * i + 1] };
      v2h t3 = { (_Float16)p3[2 * i], (_Float16)p3[2 * i + 1] };
      u0[i] = __builtin_bit_cast(unsigned, t0);
      u1[i] = __builtin_bit_cast(unsigned, t1);
      u2[i] = __builtin_bit_cast(unsigned, t2);
      u3[i] = __builtin_bit_cast(unsigned, t3);
    }
    v8u pfu0, pfu1;
    #pragma unroll
    for (int i = 0; i < 4; ++i) {
      const unsigned a0 = (unsigned)__shfl_xor((int)u0[i], 16, 32);
      const unsigned a1 = (unsigned)__shfl_xor((int)u1[i], 16, 32);
      const unsigned a2 = (unsigned)__shfl_xor((int)u2[i], 16, 32);
      const unsigned a3 = (unsigned)__shfl_xor((int)u3[i], 16, 32);
      pfu0[i]     = hi ? a1    : u0[i];   // keys hi*16 + 0..7
      pfu0[i + 4] = hi ? u1[i] : a0;      // keys hi*16 + 8..15
      pfu1[i]     = hi ? a3    : u2[i];   // keys 32 + hi*16 + 0..7
      pfu1[i + 4] = hi ? u3[i] : a2;      // keys 32 + hi*16 + 8..15
    }
    const v16h pf0 = __builtin_bit_cast(v16h, pfu0);
    const v16h pf1 = __builtin_bit_cast(v16h, pfu1);

    // ---- O^T += (mask*V)^T x P^T (4 dim-tiles x 2 key-chunks) ----
    acc[0] = __builtin_amdgcn_wmma_f32_16x16x32_f16(false, va0[0], false, pf0, (short)0, acc[0], false, false);
    acc[1] = __builtin_amdgcn_wmma_f32_16x16x32_f16(false, va0[1], false, pf0, (short)0, acc[1], false, false);
    acc[2] = __builtin_amdgcn_wmma_f32_16x16x32_f16(false, va0[2], false, pf0, (short)0, acc[2], false, false);
    acc[3] = __builtin_amdgcn_wmma_f32_16x16x32_f16(false, va0[3], false, pf0, (short)0, acc[3], false, false);
    acc[0] = __builtin_amdgcn_wmma_f32_16x16x32_f16(false, va1[0], false, pf1, (short)0, acc[0], false, false);
    acc[1] = __builtin_amdgcn_wmma_f32_16x16x32_f16(false, va1[1], false, pf1, (short)0, acc[1], false, false);
    acc[2] = __builtin_amdgcn_wmma_f32_16x16x32_f16(false, va1[2], false, pf1, (short)0, acc[2], false, false);
    acc[3] = __builtin_amdgcn_wmma_f32_16x16x32_f16(false, va1[3], false, pf1, (short)0, acc[3], false, false);
  }

  // ---- epilogue: combine partner-half sums, normalize, store ----
  l += __shfl_xor(l, 16, 32);
  const float inv = 1.f / l;
  float* orow = Og + (((size_t)bh * S_) + (size_t)qblk * QB + wave * 16 + q) * D_;
  #pragma unroll
  for (int t = 0; t < 4; ++t) {
    float4 o0 = make_float4(acc[t][0] * inv, acc[t][1] * inv, acc[t][2] * inv, acc[t][3] * inv);
    float4 o1 = make_float4(acc[t][4] * inv, acc[t][5] * inv, acc[t][6] * inv, acc[t][7] * inv);
    ((float4*)orow)[t * 4 + hi * 2 + 0] = o0;   // dims t*16 + hi*8 + 0..3
    ((float4*)orow)[t * 4 + hi * 2 + 1] = o1;   // dims t*16 + hi*8 + 4..7
  }
}

extern "C" void kernel_launch(void* const* d_in, const int* in_sizes, int n_in,
                              void* d_out, int out_size, void* d_ws, size_t ws_size,
                              hipStream_t stream) {
  (void)in_sizes; (void)n_in; (void)out_size; (void)d_ws; (void)ws_size;
  const float* Q = (const float*)d_in[0];
  const float* K = (const float*)d_in[1];
  const float* V = (const float*)d_in[2];
  const float* M = (const float*)d_in[3];
  float* O = (float*)d_out;
  dim3 grid(S_ / QB, B_ * H_);   // (16 query blocks, 48 batch*heads)
  dim3 block(256);
  fa_fwd_kernel<<<grid, block, 0, stream>>>(Q, K, V, M, O);
}
